// MultiVQLayer_25984552141245
// MI455X (gfx1250) — matrix operand order repaired and checked by reading, not compiled, not run on hardware
//
#include <hip/hip_runtime.h>
#include <stdint.h>

#define B_ 8192
#define C_ 4
#define K_ 8192
#define D_ 1024
#define BM 64
#define BETA 0.25f

// x LDS row stride in u32: 1024 + 12 pad -> byte stride 4144 (16B aligned, bank stride 12 -> conflict-free)
#define XROW 1036
#define XPK_BYTES (BM * XROW * 4)
// cb tile: 16 codes x 128 d, bf16 plane, row stride 256+16 = 272 B (16B aligned, bank stride 4)
#define CBROW 272
#define CBTILE (16 * CBROW)            // 4352 B per plane
#define CBT_BYTES (8 * CBTILE)         // [ks2][buf2][plane2]
#define MERGE_OFF (XPK_BYTES + CBT_BYTES)
#define LDS_BYTES (MERGE_OFF + 2 * 64 * 8)

typedef __attribute__((ext_vector_type(16))) __bf16 v16bf;
typedef __attribute__((ext_vector_type(8)))  float  v8f;
typedef int v4i __attribute__((vector_size(16)));

union BF16x16 { v16bf v; uint32_t u[8]; };

#if defined(__AMDGCN__) && __has_builtin(__builtin_amdgcn_global_load_async_to_lds_b128)
#define VQ_ASYNC 1
#else
#define VQ_ASYNC 0
#endif

#if VQ_ASYNC
# if __has_builtin(__builtin_amdgcn_s_wait_asynccnt)
#  define ASYNC_WAIT(n) __builtin_amdgcn_s_wait_asynccnt(n)
# else
#  define ASYNC_WAIT(n) asm volatile("s_wait_asynccnt %0" :: "n"(n) : "memory")
# endif
#else
# define ASYNC_WAIT(n) ((void)0)
#endif

template <int OFF>
__device__ __forceinline__ void async_b128(const void* g, void* l) {
#if VQ_ASYNC
  __builtin_amdgcn_global_load_async_to_lds_b128(
      (__attribute__((address_space(1))) v4i*)((const char*)g + OFF),
      (__attribute__((address_space(3))) v4i*)((char*)l + OFF), 0, 0);
#else
  uint4 v = *reinterpret_cast<const uint4*>(reinterpret_cast<const char*>(g) + OFF);
  *reinterpret_cast<uint4*>(reinterpret_cast<char*>(l) + OFF) = v;
#endif
}

__device__ __forceinline__ uint16_t bfbits(__bf16 h) {
  return __builtin_bit_cast(uint16_t, h);
}

// split f32 into bf16 hi/lo, pack hi|lo<<16
__device__ __forceinline__ uint32_t split_pack(float v) {
  __bf16 h = (__bf16)v;
  __bf16 l = (__bf16)(v - (float)h);
  return (uint32_t)bfbits(h) | ((uint32_t)bfbits(l) << 16);
}

__device__ __forceinline__ v8f vzero8() {
  v8f z = {0.f, 0.f, 0.f, 0.f, 0.f, 0.f, 0.f, 0.f};
  return z;
}

// ---------------- prep: split codebooks to bf16 hi/lo planes, exact f32 e2 ----------------
__global__ void __launch_bounds__(256)
vq_prep(const float* __restrict__ cb, __bf16* __restrict__ cbh,
        __bf16* __restrict__ cbl, float* __restrict__ e2) {
  const int row = blockIdx.x;            // c*K + k
  const int t = threadIdx.x;
  const float* src = cb + (size_t)row * D_;
  float4 v = ((const float4*)src)[t];
  __bf16 hx = (__bf16)v.x, hy = (__bf16)v.y, hz = (__bf16)v.z, hw = (__bf16)v.w;
  ushort4 H = {bfbits(hx), bfbits(hy), bfbits(hz), bfbits(hw)};
  ushort4 L = {bfbits((__bf16)(v.x - (float)hx)), bfbits((__bf16)(v.y - (float)hy)),
               bfbits((__bf16)(v.z - (float)hz)), bfbits((__bf16)(v.w - (float)hw))};
  *(ushort4*)(cbh + (size_t)row * D_ + 4 * t) = H;
  *(ushort4*)(cbl + (size_t)row * D_ + 4 * t) = L;
  float s = v.x * v.x + v.y * v.y + v.z * v.z + v.w * v.w;
  for (int m = 1; m < 32; m <<= 1) s += __shfl_xor(s, m, 32);
  __shared__ float red[8];
  if ((t & 31) == 0) red[t >> 5] = s;
  __syncthreads();
  if (t == 0) {
    float tot = 0.f;
    for (int i = 0; i < 8; ++i) tot += red[i];
    e2[row] = tot;
  }
}

// ---------------- main: fused (e2/2 - x.e) + running argmin ----------------
__global__ void __launch_bounds__(256)
vq_argmin(const float* __restrict__ x, const __bf16* __restrict__ cbh,
          const __bf16* __restrict__ cbl, const float* __restrict__ e2,
          int* __restrict__ ind) {
  extern __shared__ char smem[];
  uint32_t* xpk = (uint32_t*)smem;
  char* cbt = smem + XPK_BYTES;
  float* mgV = (float*)(smem + MERGE_OFF);
  int* mgI = (int*)(mgV + 2 * 64);

  const int t = threadIdx.x;
  const int c = blockIdx.y;
  const int b0 = blockIdx.x * BM;

  // stage x rows -> LDS as packed bf16 hi|lo
  {
    const int r = t >> 2, q = t & 3;
    const float* gx = x + ((size_t)(b0 + r) * C_ + c) * D_ + q * 256;
    uint32_t* lx = xpk + r * XROW + q * 256;
#pragma unroll 4
    for (int j = 0; j < 64; ++j) {
      float4 v = ((const float4*)gx)[j];
      uint4 o = {split_pack(v.x), split_pack(v.y), split_pack(v.z), split_pack(v.w)};
      *(uint4*)(lx + j * 4) = o;
    }
  }
  __syncthreads();

  const int wave = t >> 5, lane = t & 31;
  const int ks = wave >> 2, rg = wave & 3;
  const int ln = lane & 15, lhg = lane >> 4;

  const uint32_t* arow = xpk + (rg * 16 + ln) * XROW;   // A: row = lane&15
  const int aofs = lhg * 8;                             // A: K base 0 / 8
  const int bdq = lhg * 8;                              // B: K base 0 / 8

  v8f acc_hh = vzero8(), acc_hl = vzero8(), acc_lh = vzero8();
  float bestv[8];
  int bestk[8];
#pragma unroll
  for (int r = 0; r < 8; ++r) { bestv[r] = 3.4e38f; bestk[r] = 0; }

  const int NS = 2048;  // 256 kt-pairs * 8 stages of 128 d

  auto issue = [&](int s) {
    const int buf = s & 1;
    const int kt = 2 * (s >> 3) + ks;
    const int dst0 = (s & 7) * 128;
    const int tloc = t & 127;
    const int kk = tloc >> 3, dp = tloc & 7;
    const size_t grow = ((size_t)c * K_ + (size_t)kt * 16 + kk) * D_ + dst0 + dp * 16;
    char* dl_h = cbt + ((ks * 2 + buf) * 2 + 0) * CBTILE + kk * CBROW + dp * 32;
    char* dl_l = cbt + ((ks * 2 + buf) * 2 + 1) * CBTILE + kk * CBROW + dp * 32;
    async_b128<0>(cbh + grow, dl_h);
    async_b128<16>(cbh + grow, dl_h);
    async_b128<0>(cbl + grow, dl_l);
    async_b128<16>(cbl + grow, dl_l);
  };

  issue(0);
  for (int s = 0; s < NS; ++s) {
    if ((s & 7) == 0) { acc_hh = vzero8(); acc_hl = vzero8(); acc_lh = vzero8(); }
    if (s + 1 < NS) { issue(s + 1); ASYNC_WAIT(4); } else { ASYNC_WAIT(0); }
    __syncthreads();

    const int buf = s & 1;
    const char* th = cbt + ((ks * 2 + buf) * 2 + 0) * CBTILE + ln * CBROW;
    const char* tl = cbt + ((ks * 2 + buf) * 2 + 1) * CBTILE + ln * CBROW;
    const int dst0 = (s & 7) * 128;
#pragma unroll
    for (int dsub = 0; dsub < 4; ++dsub) {
      const uint32_t* ap = arow + dst0 + dsub * 32 + aofs;
      uint4 q0 = *(const uint4*)(ap);
      uint4 q1 = *(const uint4*)(ap + 4);
      uint4 q2 = *(const uint4*)(ap + 16);
      uint4 q3 = *(const uint4*)(ap + 20);
      uint32_t pr[16] = {q0.x, q0.y, q0.z, q0.w, q1.x, q1.y, q1.z, q1.w,
                         q2.x, q2.y, q2.z, q2.w, q3.x, q3.y, q3.z, q3.w};
      BF16x16 ah, al;
#pragma unroll
      for (int i = 0; i < 8; ++i) {
        ah.u[i] = (pr[2 * i] & 0xffffu) | (pr[2 * i + 1] << 16);
        al.u[i] = (pr[2 * i] >> 16) | (pr[2 * i + 1] & 0xffff0000u);
      }
      BF16x16 bh, bl;
      const char* ph = th + (dsub * 32 + bdq) * 2;
      const char* pl = tl + (dsub * 32 + bdq) * 2;
      *(uint4*)&bh.u[0] = *(const uint4*)(ph);
      *(uint4*)&bh.u[4] = *(const uint4*)(ph + 32);
      *(uint4*)&bl.u[0] = *(const uint4*)(pl);
      *(uint4*)&bl.u[4] = *(const uint4*)(pl + 32);
      acc_hh = __builtin_amdgcn_wmma_f32_16x16x32_bf16(false, ah.v, false, bh.v, (short)0, acc_hh, false, false);
      acc_hl = __builtin_amdgcn_wmma_f32_16x16x32_bf16(false, ah.v, false, bl.v, (short)0, acc_hl, false, false);
      acc_lh = __builtin_amdgcn_wmma_f32_16x16x32_bf16(false, al.v, false, bh.v, (short)0, acc_lh, false, false);
    }

    if ((s & 7) == 7) {  // kt finished: dist = e2/2 - xe, ascending k -> strict < keeps first-min
      const int k0 = (2 * (s >> 3) + ks) * 16;
      const float e2k = e2[(size_t)c * K_ + k0 + ln];
      const int kk = k0 + ln;
#pragma unroll
      for (int r = 0; r < 8; ++r) {
        float dv = 0.5f * e2k - (acc_hh[r] + acc_hl[r] + acc_lh[r]);
        if (dv < bestv[r]) { bestv[r] = dv; bestk[r] = kk; }
      }
    }
    __syncthreads();
  }

  // reduce across the 16 lanes holding the same row (xor of bits 0..3 stays in half-wave)
#pragma unroll
  for (int r = 0; r < 8; ++r) {
#pragma unroll
    for (int m = 1; m < 16; m <<= 1) {
      float ov = __shfl_xor(bestv[r], m, 32);
      int oi = __shfl_xor(bestk[r], m, 32);
      if (ov < bestv[r] || (ov == bestv[r] && oi < bestk[r])) { bestv[r] = ov; bestk[r] = oi; }
    }
  }
  if (ln == 0) {
#pragma unroll
    for (int r = 0; r < 8; ++r) {
      int row = rg * 16 + r + lhg * 8;
      mgV[ks * 64 + row] = bestv[r];
      mgI[ks * 64 + row] = bestk[r];
    }
  }
  __syncthreads();
  if (t < 64) {
    float v0 = mgV[t], v1 = mgV[64 + t];
    int i0 = mgI[t], i1 = mgI[64 + t];
    int best = (v1 < v0 || (v1 == v0 && i1 < i0)) ? i1 : i0;
    ind[(size_t)(b0 + t) * C_ + c] = best;
  }
}

// ---------------- gather x_q, per-row SSE (deterministic), counts, ind->float ----------------
__global__ void __launch_bounds__(256)
vq_gather(const float* __restrict__ x, const float* __restrict__ cb,
          const int* __restrict__ ind, float* __restrict__ out,
          int* __restrict__ counts, float* __restrict__ rowsse,
          float* __restrict__ indout) {
  const int row = blockIdx.x;  // b*C + c
  const int t = threadIdx.x;
  const int c = row & (C_ - 1);
  const int idx = ind[row];
  const float* xr = x + (size_t)row * D_;
  const float* qr = cb + ((size_t)c * K_ + idx) * D_;
  float4 xv = ((const float4*)xr)[t];
  float4 qv = ((const float4*)qr)[t];
  ((float4*)(out + (size_t)row * D_))[t] = qv;
  float dx = xv.x - qv.x, dy = xv.y - qv.y, dz = xv.z - qv.z, dw = xv.w - qv.w;
  float s = dx * dx + dy * dy + dz * dz + dw * dw;
  for (int m = 1; m < 32; m <<= 1) s += __shfl_xor(s, m, 32);
  __shared__ float red[8];
  if ((t & 31) == 0) red[t >> 5] = s;
  __syncthreads();
  if (t == 0) {
    float tot = 0.f;
    for (int i = 0; i < 8; ++i) tot += red[i];
    rowsse[row] = tot;
    atomicAdd(&counts[(size_t)c * K_ + idx], 1);
    indout[row] = (float)idx;
  }
}

// ---------------- finalize: loss scalar + unused code count ----------------
__global__ void __launch_bounds__(256)
vq_finalize(const int* __restrict__ counts, const float* __restrict__ rowsse,
            float* __restrict__ outScalars) {
  __shared__ float fs[256];
  __shared__ int zs[256];
  const int t = threadIdx.x;
  float s = 0.f;
  int z = 0;
  for (int i = t; i < C_ * K_; i += 256) z += (counts[i] == 0) ? 1 : 0;
  for (int i = t; i < B_ * C_; i += 256) s += rowsse[i];
  fs[t] = s; zs[t] = z;
  __syncthreads();
  for (int off = 128; off > 0; off >>= 1) {
    if (t < off) { fs[t] += fs[t + off]; zs[t] += zs[t + off]; }
    __syncthreads();
  }
  if (t == 0) {
    outScalars[0] = fs[0] * ((1.0f + BETA) / ((float)B_ * (float)D_));
    outScalars[1] = (float)zs[0];
  }
}

extern "C" void kernel_launch(void* const* d_in, const int* in_sizes, int n_in,
                              void* d_out, int out_size, void* d_ws, size_t ws_size,
                              hipStream_t stream) {
  const float* x = (const float*)d_in[0];
  const float* cb = (const float*)d_in[1];
  float* out = (float*)d_out;

  char* w = (char*)d_ws;
  const size_t CBN = (size_t)C_ * K_ * D_;          // 33554432
  __bf16* cbh = (__bf16*)w;                         // 64 MB
  __bf16* cbl = (__bf16*)(w + 2 * CBN);             // 64 MB
  char* w2 = w + 4 * CBN;                           // 128 MB
  float* e2 = (float*)(w2);                         // 128 KB
  int* ind = (int*)(w2 + 131072);                   // 128 KB
  int* counts = (int*)(w2 + 2 * 131072);            // 128 KB
  float* rowsse = (float*)(w2 + 3 * 131072);        // 128 KB

  (void)hipMemsetAsync(counts, 0, (size_t)C_ * K_ * sizeof(int), stream);

  vq_prep<<<C_ * K_, 256, 0, stream>>>(cb, cbh, cbl, e2);

  (void)hipFuncSetAttribute((const void*)vq_argmin,
                            hipFuncAttributeMaxDynamicSharedMemorySize, (int)LDS_BYTES);
  vq_argmin<<<dim3(B_ / BM, C_), 256, LDS_BYTES, stream>>>(x, cbh, cbl, e2, ind);

  const size_t XQ = (size_t)B_ * C_ * D_;
  vq_gather<<<B_ * C_, 256, 0, stream>>>(x, cb, ind, out, counts, rowsse, out + XQ + 2);
  vq_finalize<<<1, 256, 0, stream>>>(counts, rowsse, out + XQ);
}
